// LSTM_1348619731216
// MI455X (gfx1250) — matrix operand order repaired and checked by reading, not compiled, not run on hardware
//
#include <hip/hip_runtime.h>
#include <hip/hip_bf16.h>

// ---------------------------------------------------------------------------
// Persistent fused 2-layer LSTM + FC head for MI455X (gfx1250, wave32, WMMA).
//
// Batch rows are recurrence-independent -> each block owns 16 rows for all
// 365 timesteps. Hidden state h lives in LDS (bf16, WMMA A-operand layout);
// cell state c lives entirely in registers (lane <-> (row, hidden) is static).
// 16 waves: wave = gate N-tile for the GEMM phases (13 real tiles cover
// 4H=200), wave = batch row for the pointwise phases. Weight B-fragments are
// register-resident. Activations use hardware v_tanh_f32 (TRANS pipe).
// x staging is software-pipelined one step ahead; FC head is a wave-level
// __shfl_xor reduction. Global traffic ~ read x (27MB) + out (3MB).
// ---------------------------------------------------------------------------

#define T_STEPS 365
#define BATCH   2048
#define INSZ    9
#define HSZ     50
#define G4      200     // 4*H
#define KPAD    64      // hidden padded for K
#define MTILE   16      // batch rows per block
#define THREADS 512     // 16 wave32s
#define NT_REAL 13      // ceil(200/16): waves >= 13 have all-zero gate tiles
#define GSTRIDE 256     // gates LDS row stride (f32)

typedef __attribute__((ext_vector_type(16))) __bf16 v16bf;
typedef __attribute__((ext_vector_type(8)))  float  v8f;

union Frag16 {
    unsigned int   u32[8];
    unsigned short us[16];
    v16bf          v;
};

static __device__ __forceinline__ unsigned short f2bf(float f) {
    unsigned int u = __float_as_uint(f);
    unsigned int r = u + 0x7FFFu + ((u >> 16) & 1u);   // round-to-nearest-even
    return (unsigned short)(r >> 16);
}

static __device__ __forceinline__ float fast_rcp(float x) {
#if __has_builtin(__builtin_amdgcn_rcpf)
    return __builtin_amdgcn_rcpf(x);
#else
    return 1.0f / x;
#endif
}

static __device__ __forceinline__ float fast_tanh(float x) {
#if __has_builtin(__builtin_amdgcn_tanhf)
    return __builtin_amdgcn_tanhf(x);          // v_tanh_f32 (TRANS pipe)
#else
    float e = __expf(2.0f * x);
    return (e - 1.0f) * fast_rcp(e + 1.0f);
#endif
}

static __device__ __forceinline__ float sigmoidf_(float x) {
    // sigmoid(x) = 0.5*tanh(x/2) + 0.5  -> 1 TRANS + 2 VALU, no division
    return fmaf(fast_tanh(0.5f * x), 0.5f, 0.5f);
}

// A-fragment (16x32 bf16, MxK) from LDS, rows of `rowStrideHalf` ushorts.
// ISA 7.12.2: lane l -> M = l%16, K-base = 8*(l/16); element i ->
// K = base + ((i>=8)?16:0) + (i&7). Pairs are consecutive K -> b32 reads
// (compiler merges into ds_load_b128).
static __device__ __forceinline__ v16bf load_a_frag(const unsigned short* base,
                                                    int rowStrideHalf,
                                                    int lane, int kk) {
    Frag16 f;
    const unsigned short* row = base + (lane & 15) * rowStrideHalf;
    int kb = kk + ((lane >> 4) << 3);
#pragma unroll
    for (int p = 0; p < 8; ++p) {
        int k0 = kb + ((p >= 4) ? 16 : 0) + ((p & 3) << 1);
        f.u32[p] = *(const unsigned int*)(row + k0);
    }
    return f.v;
}

// B-fragment (32x16 bf16, KxN) straight from global f32 weights W[Nreal][Kreal]
// (PyTorch w_* are [4H, K] row-major == our [n][k]).
// Layout: lane l -> N = l%16, K = kb + 8*(l/16) + ((i>=8)?16:0) + (i&7). Zero-padded.
static __device__ __forceinline__ v16bf load_b_frag(const float* __restrict__ W,
                                                    int Nreal, int Kreal,
                                                    int lane, int nt, int kb) {
    Frag16 f;
    int n  = nt * 16 + (lane & 15);
    int kh = kb + ((lane >> 4) << 3);
#pragma unroll
    for (int i = 0; i < 16; ++i) {
        int k = kh + i + ((i >= 8) ? 8 : 0);
        float v = (n < Nreal && k < Kreal) ? W[n * Kreal + k] : 0.0f;
        f.us[i] = f2bf(v);
    }
    return f.v;
}

static __device__ __forceinline__ v8f wmma_bf16(v16bf a, v16bf b, v8f c) {
    return __builtin_amdgcn_wmma_f32_16x16x32_bf16(
        /*neg_a=*/false, a, /*neg_b=*/false, b,
        /*c_mod=*/(short)0, c, /*reuse_a=*/false, /*reuse_b=*/false);
}

__global__ void __launch_bounds__(THREADS)
lstm2_persistent(const float* __restrict__ x,
                 const float* __restrict__ w_ih0, const float* __restrict__ w_hh0,
                 const float* __restrict__ b_ih0, const float* __restrict__ b_hh0,
                 const float* __restrict__ w_ih1, const float* __restrict__ w_hh1,
                 const float* __restrict__ b_ih1, const float* __restrict__ b_hh1,
                 const float* __restrict__ fc_w,  const float* __restrict__ fc_b,
                 float* __restrict__ out)
{
    __shared__ alignas(16) unsigned short s_x [MTILE][32];    // x_t tile (bf16, K-padded)
    __shared__ alignas(16) unsigned short s_hA[MTILE][KPAD];  // layer0 hidden (bf16)
    __shared__ alignas(16) unsigned short s_hB[MTILE][KPAD];  // layer1 hidden (bf16)
    __shared__ alignas(16) float s_gates[MTILE][GSTRIDE];     // 16 KB

    const int tid   = threadIdx.x;
    const int lane  = tid & 31;
    const int wave  = tid >> 5;     // GEMM: N-tile index; pointwise: batch row
    const int bbase = blockIdx.x * MTILE;

    // ---- zero-init LDS (bf16 bits 0 == 0.0) ----------------------------------
    for (int i = tid; i < MTILE * 32;   i += THREADS) (&s_x [0][0])[i] = 0;
    for (int i = tid; i < MTILE * KPAD; i += THREADS) (&s_hA[0][0])[i] = 0;
    for (int i = tid; i < MTILE * KPAD; i += THREADS) (&s_hB[0][0])[i] = 0;

    // ---- preload weight B-fragments into registers (one N-tile per wave) -----
    const int nt = wave;
    v16bf Bih0   = load_b_frag(w_ih0, G4, INSZ, lane, nt, 0);   // K=9 in one K32 frag
    v16bf Bhh0_0 = load_b_frag(w_hh0, G4, HSZ,  lane, nt, 0);
    v16bf Bhh0_1 = load_b_frag(w_hh0, G4, HSZ,  lane, nt, 32);
    v16bf Bih1_0 = load_b_frag(w_ih1, G4, HSZ,  lane, nt, 0);
    v16bf Bih1_1 = load_b_frag(w_ih1, G4, HSZ,  lane, nt, 32);
    v16bf Bhh1_0 = load_b_frag(w_hh1, G4, HSZ,  lane, nt, 0);
    v16bf Bhh1_1 = load_b_frag(w_hh1, G4, HSZ,  lane, nt, 32);
    const int   ncol  = nt * 16 + (lane & 15);
    const float bias0 = (ncol < G4) ? (b_ih0[ncol] + b_hh0[ncol]) : 0.0f;
    const float bias1 = (ncol < G4) ? (b_ih1[ncol] + b_hh1[ncol]) : 0.0f;

    // FC weights: lane j owns fc_w[j] and fc_w[j+32]; bias broadcast
    const float fcw0 = fc_w[lane];                           // lane < 32 < 50
    const float fcw1 = (lane + 32 < HSZ) ? fc_w[lane + 32] : 0.0f;
    const float fcb  = fc_b[0];

    // cell state lives in registers: this lane's (row=wave, jj=lane[, lane+32])
    float cA0 = 0.0f, cA1 = 0.0f, cB0 = 0.0f, cB1 = 0.0f;

    const bool gemm_wave = (wave < NT_REAL);     // wave-uniform branch
    const bool stager    = (tid < MTILE * INSZ); // 144 threads stage x tiles
    const int  sm  = tid / INSZ, sii = tid - (tid / INSZ) * INSZ;
    const bool lane2 = (lane < HSZ - 32);        // lanes 0..17 own jj+32

    __syncthreads();
    // ---- prologue: stage x(t=0) ----------------------------------------------
    if (stager)
        s_x[sm][sii] = f2bf(x[((size_t)bbase + sm) * INSZ + sii]);
    __syncthreads();

    // ---- time loop (4 barriers per step) -------------------------------------
    for (int t = 0; t < T_STEPS; ++t) {
        // software-pipelined x staging: issue load for t+1 now, commit later
        float xnext;
        const bool do_stage = stager && (t + 1 < T_STEPS);
        if (do_stage)
            xnext = x[((size_t)(t + 1) * BATCH + bbase + sm) * INSZ + sii];
        if (t + 2 < T_STEPS && tid >= THREADS - MTILE) {
            int m = tid - (THREADS - MTILE);
            __builtin_prefetch(&x[((size_t)(t + 2) * BATCH + bbase + m) * INSZ], 0, 0);
        }

        // ---- layer 0: gates = x@Wih0^T + h0@Whh0^T + bias (WMMA) ------------
        v16bf aS0, aS1;   // layer-1 state frags: stable since last ew1, hoisted
        if (gemm_wave) {
            aS0 = load_a_frag(&s_hB[0][0], KPAD, lane, 0);
            aS1 = load_a_frag(&s_hB[0][0], KPAD, lane, 32);
            v16bf aX  = load_a_frag(&s_x [0][0], 32,   lane, 0);
            v16bf aH0 = load_a_frag(&s_hA[0][0], KPAD, lane, 0);
            v16bf aH1 = load_a_frag(&s_hA[0][0], KPAD, lane, 32);
            v8f acc = {};
            acc = wmma_bf16(aX,  Bih0,   acc);
            acc = wmma_bf16(aH0, Bhh0_0, acc);
            acc = wmma_bf16(aH1, Bhh0_1, acc);
            int mb = (lane >> 4) * 8;
#pragma unroll
            for (int r = 0; r < 8; ++r)
                s_gates[mb + r][ncol] = acc[r] + bias0;
        }
        __syncthreads();

        // commit x(t+1): s_x only read by gemm0, which is behind the barrier
        if (do_stage)
            s_x[sm][sii] = f2bf(xnext);

        // ---- layer 0 pointwise: wave = row, lane = hidden index -------------
        {
            const int m = wave;
            float gi = sigmoidf_(s_gates[m][lane]);
            float gf = sigmoidf_(s_gates[m][HSZ + lane]);
            float gg = fast_tanh(s_gates[m][2 * HSZ + lane]);
            float go = sigmoidf_(s_gates[m][3 * HSZ + lane]);
            cA0 = fmaf(gf, cA0, gi * gg);
            s_hA[m][lane] = f2bf(go * fast_tanh(cA0));
            if (lane2) {
                const int jj = lane + 32;
                float gi2 = sigmoidf_(s_gates[m][jj]);
                float gf2 = sigmoidf_(s_gates[m][HSZ + jj]);
                float gg2 = fast_tanh(s_gates[m][2 * HSZ + jj]);
                float go2 = sigmoidf_(s_gates[m][3 * HSZ + jj]);
                cA1 = fmaf(gf2, cA1, gi2 * gg2);
                s_hA[m][jj] = f2bf(go2 * fast_tanh(cA1));
            }
        }
        __syncthreads();

        // ---- layer 1: gates = h0'@Wih1^T + h1@Whh1^T + bias (WMMA) ----------
        if (gemm_wave) {
            v16bf aI0 = load_a_frag(&s_hA[0][0], KPAD, lane, 0);
            v16bf aI1 = load_a_frag(&s_hA[0][0], KPAD, lane, 32);
            v8f acc = {};
            acc = wmma_bf16(aI0, Bih1_0, acc);
            acc = wmma_bf16(aI1, Bih1_1, acc);
            acc = wmma_bf16(aS0, Bhh1_0, acc);
            acc = wmma_bf16(aS1, Bhh1_1, acc);
            int mb = (lane >> 4) * 8;
#pragma unroll
            for (int r = 0; r < 8; ++r)
                s_gates[mb + r][ncol] = acc[r] + bias1;
        }
        __syncthreads();

        // ---- layer 1 pointwise + FC head (wave shuffle reduction) -----------
        {
            const int m = wave;
            float gi = sigmoidf_(s_gates[m][lane]);
            float gf = sigmoidf_(s_gates[m][HSZ + lane]);
            float gg = fast_tanh(s_gates[m][2 * HSZ + lane]);
            float go = sigmoidf_(s_gates[m][3 * HSZ + lane]);
            cB0 = fmaf(gf, cB0, gi * gg);
            float h0 = go * fast_tanh(cB0);
            s_hB[m][lane] = f2bf(h0);
            float p = h0 * fcw0;
            if (lane2) {
                const int jj = lane + 32;
                float gi2 = sigmoidf_(s_gates[m][jj]);
                float gf2 = sigmoidf_(s_gates[m][HSZ + jj]);
                float gg2 = fast_tanh(s_gates[m][2 * HSZ + jj]);
                float go2 = sigmoidf_(s_gates[m][3 * HSZ + jj]);
                cB1 = fmaf(gf2, cB1, gi2 * gg2);
                float h1 = go2 * fast_tanh(cB1);
                s_hB[m][jj] = f2bf(h1);
                p = fmaf(h1, fcw1, p);
            }
#pragma unroll
            for (int s = 16; s >= 1; s >>= 1)
                p += __shfl_xor(p, s, 32);
            if (lane == 0)
                out[(size_t)t * BATCH + bbase + m] = p + fcb;
        }
        __syncthreads();   // orders ew1 gate reads before next step's gate writes
    }
}

extern "C" void kernel_launch(void* const* d_in, const int* in_sizes, int n_in,
                              void* d_out, int out_size, void* d_ws, size_t ws_size,
                              hipStream_t stream) {
    const float* x     = (const float*)d_in[0];
    const float* w_ih0 = (const float*)d_in[1];
    const float* w_hh0 = (const float*)d_in[2];
    const float* b_ih0 = (const float*)d_in[3];
    const float* b_hh0 = (const float*)d_in[4];
    const float* w_ih1 = (const float*)d_in[5];
    const float* w_hh1 = (const float*)d_in[6];
    const float* b_ih1 = (const float*)d_in[7];
    const float* b_hh1 = (const float*)d_in[8];
    const float* fc_w  = (const float*)d_in[9];
    const float* fc_b  = (const float*)d_in[10];
    float* out = (float*)d_out;

    dim3 grid(BATCH / MTILE);   // 128 independent batch tiles
    dim3 block(THREADS);        // 16 wave32s
    hipLaunchKernelGGL(lstm2_persistent, grid, block, 0, stream,
                       x, w_ih0, w_hh0, b_ih0, b_hh0,
                       w_ih1, w_hh1, b_ih1, b_hh1, fc_w, fc_b, out);
}
